// MLA_66279935312321
// MI455X (gfx1250) — compile-verified, matrix operand order checked by reading, not run on hardware
//
#include <hip/hip_runtime.h>
#include <hip/hip_bf16.h>

// ---------------------------------------------------------------------------
// MLA forward for gfx1250 (MI455X). All dense contractions run on
// v_wmma_f32_16x16x32_bf16 (wave32 WMMA, f32 accumulate). Attention is a
// flash-style kernel with TDM (tensor_load_to_lds) staged K tiles and
// online softmax.
// ---------------------------------------------------------------------------

typedef __attribute__((ext_vector_type(16))) __bf16 bf16x16;
typedef __attribute__((ext_vector_type(8)))  __bf16 bf16x8;
typedef __attribute__((ext_vector_type(8)))  float  f32x8;
typedef __attribute__((ext_vector_type(4)))  unsigned int u32x4;
typedef __attribute__((ext_vector_type(4)))  int    i32x4;
typedef __attribute__((ext_vector_type(8)))  int    i32x8;

#define MLA_DIM     2048
#define MLA_NH      16
#define MLA_QLORA   1536
#define MLA_KVLORA  512
#define MLA_NOPE    128
#define MLA_ROPE    64
#define MLA_VHEAD   128
#define MLA_QKHEAD  192
#define MLA_B       2
#define MLA_S       2048
#define MLA_BS      (MLA_B * MLA_S)           // 4096 token rows
#define MLA_KD      (MLA_KVLORA + MLA_ROPE)   // 576 effective K dim
#define MLA_SCALE   0.07216878364870322f      // 192^-0.5
#define ATT_TT      32                        // t-tile for flash attention

// ---------------------------------------------------------------------------
// f32 -> bf16 elementwise convert
// ---------------------------------------------------------------------------
__global__ void mla_cvt_bf16(const float* __restrict__ in,
                             __bf16* __restrict__ out, long n) {
  long i = (long)blockIdx.x * blockDim.x + threadIdx.x;
  long stride = (long)gridDim.x * blockDim.x;
  for (; i < n; i += stride) out[i] = (__bf16)in[i];
}

// ---------------------------------------------------------------------------
// RMSNorm: one block per row; f32 in (row stride in_stride), bf16 out.
// ---------------------------------------------------------------------------
__global__ void mla_rmsnorm(const float* __restrict__ in, long in_stride,
                            const float* __restrict__ w,
                            __bf16* __restrict__ out, long out_stride, int L) {
  const int row = blockIdx.x, tid = threadIdx.x;
  const float* x = in + (long)row * in_stride;
  __shared__ float red[256];
  float ss = 0.f;
  for (int i = tid; i < L; i += 256) { float v = x[i]; ss += v * v; }
  red[tid] = ss;
  __syncthreads();
  for (int s = 128; s > 0; s >>= 1) {
    if (tid < s) red[tid] += red[tid + s];
    __syncthreads();
  }
  const float r = rsqrtf(red[0] / (float)L + 1e-6f);
  __bf16* o = out + (long)row * out_stride;
  for (int i = tid; i < L; i += 256) o[i] = (__bf16)(x[i] * r * w[i]);
}

// ---------------------------------------------------------------------------
// WMMA GEMM: C[M,N] = A[M,K] (bf16, row stride lda) x B^T,
// B[n,k] = Bp[n*ldbn + k*ldbk]; templated on BK1 (k-contiguous B).
// Each wave produces a 16x64 strip: the A operand is loaded once per K-step
// and reused across 4 WMMAs (4 f32x8 accumulators).
// Per-lane operand layout (ISA 7.12.2, 16-bit A/B 16x32):
//   lane&15 = M (or N) index; lane>>4 selects K-half {0..7,16..23}/{8..15,24..31}
// ---------------------------------------------------------------------------
template <bool BK1>
__global__ __launch_bounds__(256) void mla_wmma_gemm_t(
    const __bf16* __restrict__ A, long lda,
    const __bf16* __restrict__ B, long ldbn, long ldbk,
    float* __restrict__ Cf, __bf16* __restrict__ Cb, long ldc,
    int M, int N, int K) {
  const int tiles_n = N >> 6;  // 64-wide wave strips
  const long tiles = (long)(M >> 4) * tiles_n;
  const long wid = (long)blockIdx.x * (blockDim.x >> 5) + (threadIdx.x >> 5);
  if (wid >= tiles) return;
  const int tm = (int)(wid / tiles_n);
  const int tn = (int)(wid % tiles_n);
  const int lane = threadIdx.x & 31;
  const int rowl = lane & 15;
  const int half = lane >> 4;
  const int kb = half * 8;

  const __bf16* ap = A + (long)(tm * 16 + rowl) * lda + kb;

  f32x8 acc[4];
  for (int j = 0; j < 4; ++j)
    for (int i = 0; i < 8; ++i) acc[j][i] = 0.f;

  if (BK1) {
    const __bf16* bp0 = B + (long)(tn * 64 + rowl) * ldbn + kb;
    const __bf16* bp1 = bp0 + 16 * ldbn;
    const __bf16* bp2 = bp0 + 32 * ldbn;
    const __bf16* bp3 = bp0 + 48 * ldbn;
    for (int k0 = 0; k0 < K; k0 += 32) {
      union { bf16x16 v; bf16x8 h8[2]; } a, b0, b1, b2, b3;
      a.h8[0] = *(const bf16x8*)(ap);
      a.h8[1] = *(const bf16x8*)(ap + 16);
      ap += 32;
      b0.h8[0] = *(const bf16x8*)(bp0);
      b0.h8[1] = *(const bf16x8*)(bp0 + 16);
      bp0 += 32;
      b1.h8[0] = *(const bf16x8*)(bp1);
      b1.h8[1] = *(const bf16x8*)(bp1 + 16);
      bp1 += 32;
      b2.h8[0] = *(const bf16x8*)(bp2);
      b2.h8[1] = *(const bf16x8*)(bp2 + 16);
      bp2 += 32;
      b3.h8[0] = *(const bf16x8*)(bp3);
      b3.h8[1] = *(const bf16x8*)(bp3 + 16);
      bp3 += 32;
      acc[0] = __builtin_amdgcn_wmma_f32_16x16x32_bf16(false, a.v, false, b0.v, (short)0, acc[0], false, false);
      acc[1] = __builtin_amdgcn_wmma_f32_16x16x32_bf16(false, a.v, false, b1.v, (short)0, acc[1], false, false);
      acc[2] = __builtin_amdgcn_wmma_f32_16x16x32_bf16(false, a.v, false, b2.v, (short)0, acc[2], false, false);
      acc[3] = __builtin_amdgcn_wmma_f32_16x16x32_bf16(false, a.v, false, b3.v, (short)0, acc[3], false, false);
    }
  } else {  // B stored [K,N]: strided gather along k
    for (int k0 = 0; k0 < K; k0 += 32) {
      union { bf16x16 v; bf16x8 h8[2]; } a;
      a.h8[0] = *(const bf16x8*)(ap);
      a.h8[1] = *(const bf16x8*)(ap + 16);
      ap += 32;
      for (int j = 0; j < 4; ++j) {
        const __bf16* bp = B + (long)(tn * 64 + j * 16 + rowl) * ldbn +
                           (long)(k0 + kb) * ldbk;
        bf16x16 bv;
        for (int i = 0; i < 8; ++i) {
          bv[i]     = bp[(long)i * ldbk];
          bv[i + 8] = bp[(long)(i + 16) * ldbk];
        }
        acc[j] = __builtin_amdgcn_wmma_f32_16x16x32_bf16(false, a.v, false, bv, (short)0, acc[j], false, false);
      }
    }
  }

  const int r0 = tm * 16 + half * 8;
  for (int j = 0; j < 4; ++j) {
    const int col = tn * 64 + j * 16 + rowl;
    for (int i = 0; i < 8; ++i) {
      const long off = (long)(r0 + i) * ldc + col;
      if (Cf) Cf[off] = acc[j][i];
      else    Cb[off] = (__bf16)acc[j][i];
    }
  }
}

// ---------------------------------------------------------------------------
// Split q into q_nope (bf16) and rope(q_pe) (bf16).
// q layout: [bs][h*192 + {0..127 nope | 128..191 pe}]
// ---------------------------------------------------------------------------
__global__ void mla_qsplit_rope(const float* __restrict__ qf,
                                const float* __restrict__ fcos,
                                const float* __restrict__ fsin,
                                __bf16* __restrict__ qnope,
                                __bf16* __restrict__ qpe) {
  const long bs = blockIdx.x;
  const int tid = threadIdx.x;
  const float* qr = qf + bs * (MLA_NH * MLA_QKHEAD);
  for (int e = tid; e < MLA_NH * MLA_NOPE; e += 256) {
    const int hh = e >> 7, d = e & 127;
    qnope[bs * (MLA_NH * MLA_NOPE) + e] = (__bf16)qr[hh * MLA_QKHEAD + d];
  }
  const int s = (int)(bs & (MLA_S - 1));
  for (int p = tid; p < MLA_NH * (MLA_ROPE / 2); p += 256) {
    const int hh = p >> 5, j = p & 31;
    const float x0 = qr[hh * MLA_QKHEAD + MLA_NOPE + 2 * j];
    const float x1 = qr[hh * MLA_QKHEAD + MLA_NOPE + 2 * j + 1];
    const float c = fcos[s * 32 + j], sn = fsin[s * 32 + j];
    qpe[bs * (MLA_NH * MLA_ROPE) + hh * MLA_ROPE + 2 * j]     = (__bf16)(x0 * c - x1 * sn);
    qpe[bs * (MLA_NH * MLA_ROPE) + hh * MLA_ROPE + 2 * j + 1] = (__bf16)(x0 * sn + x1 * c);
  }
}

// rope on k_pe: kv[:, 512:576] -> bf16 [bs][64]
__global__ void mla_krope(const float* __restrict__ kvf,
                          const float* __restrict__ fcos,
                          const float* __restrict__ fsin,
                          __bf16* __restrict__ kpe) {
  const long bs = blockIdx.x;
  const int j = threadIdx.x;  // 32 threads = 32 pairs
  const int s = (int)(bs & (MLA_S - 1));
  const float x0 = kvf[bs * MLA_KD + MLA_KVLORA + 2 * j];
  const float x1 = kvf[bs * MLA_KD + MLA_KVLORA + 2 * j + 1];
  const float c = fcos[s * 32 + j], sn = fsin[s * 32 + j];
  kpe[bs * MLA_ROPE + 2 * j]     = (__bf16)(x0 * c - x1 * sn);
  kpe[bs * MLA_ROPE + 2 * j + 1] = (__bf16)(x0 * sn + x1 * c);
}

// ---------------------------------------------------------------------------
// Flash attention over the latent cache.
// Block = 256 threads (8 waves) handles one (b, h, 16-row s-tile).
// Effective Q/K rows are [q_abs|q_pe] / [kv_cache|k_pe], dim 576.
// The 32x512 kv_cache tile is DMA'd into LDS by the Tensor Data Mover with
// row padding to the 576-element pitch; k_pe columns are staged by threads.
// Waves 0-1 build the 16x32 score tile (18 WMMA K-steps); all 8 waves
// accumulate P@V with one WMMA per 16x16 output tile (c split 64/wave).
// LDS ~58KB: Q 18K + K-tile 36K + S 2K + P 1K.
// ---------------------------------------------------------------------------
__global__ __launch_bounds__(256) void mla_flash_attn(
    const __bf16* __restrict__ qabs,   // [bs][h*512 + c]
    const __bf16* __restrict__ qpe,    // [bs][h*64 + r]
    const __bf16* __restrict__ kvc,    // [bs][512]
    const __bf16* __restrict__ kpe,    // [bs][64]
    const float*  __restrict__ mask,   // [S][S]
    __bf16* __restrict__ octx) {       // [bs][h*512 + c]
  const int s_tile = blockIdx.x, h = blockIdx.y, b = blockIdx.z;
  const int tid = threadIdx.x, lane = tid & 31, wave = tid >> 5;
  const int rowl = lane & 15, half = lane >> 4, kb = half * 8;
  const int s0 = s_tile * 16;
  const long bs0 = (long)b * MLA_S + s0;

  __shared__ __bf16 Qs[16 * MLA_KD];
  __shared__ __bf16 Ks[ATT_TT * MLA_KD];
  __shared__ float  Ss[16 * ATT_TT];
  __shared__ __bf16 Ps[16 * ATT_TT];
  __shared__ float  m_run[16], l_run[16], alpha_s[16], mnew_s[16];

  // stage Q (c-latent + roped pe) once
  for (int e = tid; e < 16 * MLA_KD; e += 256) {
    const int r = e / MLA_KD, c = e % MLA_KD;
    __bf16 v;
    if (c < MLA_KVLORA)
      v = qabs[(bs0 + r) * (MLA_NH * MLA_KVLORA) + (long)h * MLA_KVLORA + c];
    else
      v = qpe[(bs0 + r) * (MLA_NH * MLA_ROPE) + (long)h * MLA_ROPE + (c - MLA_KVLORA)];
    Qs[e] = v;
  }
  if (tid < 16) { m_run[tid] = -1e30f; l_run[tid] = 0.f; }

  f32x8 acc[4];
  for (int j = 0; j < 4; ++j)
    for (int i = 0; i < 8; ++i) acc[j][i] = 0.f;

  const int ntt = s_tile / 2 + 1;  // causal: covers t <= s0+15
  for (int tt = 0; tt < ntt; ++tt) {
    const int t0 = tt * ATT_TT;
    __syncthreads();  // previous iteration done reading Ks

#if defined(__gfx1250__)
    // --- TDM: DMA 32x512 bf16 kv_cache tile -> Ks with row pad 512->576 ---
    if (wave == 0) {
      const unsigned long long gaddr =
          (unsigned long long)(const void*)(kvc + ((long)b * MLA_S + t0) * MLA_KVLORA);
      const unsigned ldsa = (unsigned)(unsigned long long)(void*)&Ks[0];
      u32x4 g0;
      g0[0] = 1u;                                   // count=1 (user descriptor)
      g0[1] = ldsa;                                 // lds_addr
      g0[2] = (unsigned)(gaddr & 0xffffffffu);      // global_addr[31:0]
      g0[3] = (unsigned)((gaddr >> 32) & 0x01ffffffu) | (2u << 30);  // type=2
      i32x8 g1;
      g1[0] = (1 << 16)     // data_size = 2 bytes
            | (1 << 20)     // pad_enable
            | (7 << 22)     // pad_interval: 256 DWORDs (one 512-bf16 row)
            | (31 << 25);   // pad_amount: 32 DWORDs (64 bf16 -> pitch 576)
      g1[1] = (int)(512u << 16);   // tensor_dim0 = 512 -> bits[79:48] low half
      g1[2] = (int)((unsigned)MLA_BS << 16);  // tensor_dim1 -> bits[111:80] low half
      g1[3] = (int)(512u << 16);   // tile_dim0 = 512 -> bits[127:112]
      g1[4] = ATT_TT;              // tile_dim1 = 32 -> bits[143:128]
      g1[5] = 512;                 // tensor_dim0_stride -> bits[207:160] low 32
      g1[6] = 0;
      g1[7] = 0;
      i32x4 gz;
      gz[0] = gz[1] = gz[2] = gz[3] = 0;
#if __clang_major__ >= 23
      i32x8 gz8;
      for (int i = 0; i < 8; ++i) gz8[i] = 0;
      __builtin_amdgcn_tensor_load_to_lds(g0, g1, gz, gz, gz8, 0);
#else
      __builtin_amdgcn_tensor_load_to_lds(g0, g1, gz, gz, 0);
#endif
    }
    // k_pe columns staged by all threads while the TDM runs
    for (int e = tid; e < ATT_TT * MLA_ROPE; e += 256) {
      const int r = e >> 6, c = e & 63;
      Ks[r * MLA_KD + MLA_KVLORA + c] =
          kpe[((long)b * MLA_S + t0 + r) * MLA_ROPE + c];
    }
    if (wave == 0) __builtin_amdgcn_s_wait_tensorcnt((short)0);
    __syncthreads();
#else
    for (int e = tid; e < ATT_TT * MLA_KD; e += 256) {
      const int r = e / MLA_KD, c = e % MLA_KD;
      const long t = (long)b * MLA_S + t0 + r;
      Ks[e] = (c < MLA_KVLORA) ? kvc[t * MLA_KVLORA + c]
                               : kpe[t * MLA_ROPE + (c - MLA_KVLORA)];
    }
    __syncthreads();
#endif

    // --- scores: Q(16x576) . K^T(576x32), waves 0 & 1 own one n-tile each
    if (wave < 2) {
      union { bf16x16 v; bf16x8 h8[2]; } a, bb;
      f32x8 sacc;
      for (int i = 0; i < 8; ++i) sacc[i] = 0.f;
      for (int k0 = 0; k0 < MLA_KD; k0 += 32) {
        a.h8[0]  = *(const bf16x8*)(Qs + rowl * MLA_KD + k0 + kb);
        a.h8[1]  = *(const bf16x8*)(Qs + rowl * MLA_KD + k0 + kb + 16);
        bb.h8[0] = *(const bf16x8*)(Ks + (wave * 16 + rowl) * MLA_KD + k0 + kb);
        bb.h8[1] = *(const bf16x8*)(Ks + (wave * 16 + rowl) * MLA_KD + k0 + kb + 16);
        sacc = __builtin_amdgcn_wmma_f32_16x16x32_bf16(
            false, a.v, false, bb.v, (short)0, sacc, false, false);
      }
      const int colt = wave * 16 + rowl;
      for (int i = 0; i < 8; ++i) {
        const int r = i + half * 8;
        Ss[r * ATT_TT + colt] =
            sacc[i] * MLA_SCALE + mask[(long)(s0 + r) * MLA_S + (t0 + colt)];
      }
    }
    __syncthreads();

    // --- online softmax bookkeeping (one thread per row)
    if (tid < 16) {
      float tm = -1e30f;
      for (int c = 0; c < ATT_TT; ++c) tm = fmaxf(tm, Ss[tid * ATT_TT + c]);
      const float mn = fmaxf(m_run[tid], tm);
      mnew_s[tid] = mn;
      alpha_s[tid] = __expf(m_run[tid] - mn);
    }
    __syncthreads();
    for (int e = tid; e < 16 * ATT_TT; e += 256) {
      const int r = e / ATT_TT;
      const float p = __expf(Ss[e] - mnew_s[r]);
      Ss[e] = p;
      Ps[e] = (__bf16)p;
    }
    __syncthreads();
    if (tid < 16) {
      float sum = 0.f;
      for (int c = 0; c < ATT_TT; ++c) sum += Ss[tid * ATT_TT + c];
      l_run[tid] = l_run[tid] * alpha_s[tid] + sum;
      m_run[tid] = mnew_s[tid];
    }
    __syncthreads();

    // --- O += P(16x32) @ V(32x512); each wave owns a 64-wide c chunk
    union { bf16x16 v; bf16x8 h8[2]; } ap;
    ap.h8[0] = *(const bf16x8*)(Ps + rowl * ATT_TT + kb);
    ap.h8[1] = *(const bf16x8*)(Ps + rowl * ATT_TT + kb + 16);
    for (int j = 0; j < 4; ++j) {
      const int c0 = wave * 64 + j * 16;
      bf16x16 bv;
      for (int i = 0; i < 8; ++i) {
        bv[i]     = Ks[(kb + i) * MLA_KD + c0 + rowl];
        bv[i + 8] = Ks[(kb + 16 + i) * MLA_KD + c0 + rowl];
      }
      f32x8 t = acc[j];
      for (int i = 0; i < 8; ++i) t[i] *= alpha_s[i + half * 8];
      acc[j] = __builtin_amdgcn_wmma_f32_16x16x32_bf16(
          false, ap.v, false, bv, (short)0, t, false, false);
    }
  }
  __syncthreads();

  // finalize: divide by l, store bf16 context
  for (int j = 0; j < 4; ++j) {
    const int c0 = wave * 64 + j * 16;
    for (int i = 0; i < 8; ++i) {
      const int r = i + half * 8;
      const float o = acc[j][i] / l_run[r];
      octx[(bs0 + r) * (MLA_NH * MLA_KVLORA) + (long)h * MLA_KVLORA + c0 + rowl] =
          (__bf16)o;
    }
  }
}

// ---------------------------------------------------------------------------
// Host orchestration
// ---------------------------------------------------------------------------
extern "C" void kernel_launch(void* const* d_in, const int* in_sizes, int n_in,
                              void* d_out, int out_size, void* d_ws, size_t ws_size,
                              hipStream_t stream) {
  (void)in_sizes; (void)n_in; (void)out_size; (void)ws_size;
  const float* x         = (const float*)d_in[0];
  const float* freqs_cos = (const float*)d_in[1];
  const float* freqs_sin = (const float*)d_in[2];
  const float* mask      = (const float*)d_in[3];
  const float* wq_a      = (const float*)d_in[4];
  const float* q_norm_w  = (const float*)d_in[5];
  const float* wq_b      = (const float*)d_in[6];
  const float* wkv_a     = (const float*)d_in[7];
  const float* kv_norm_w = (const float*)d_in[8];
  const float* wkv_b     = (const float*)d_in[9];
  const float* wo        = (const float*)d_in[10];
  float* out = (float*)d_out;

  // ---- workspace arena (~326 MB) ----
  char* wp = (char*)d_ws;
  auto arena = [&](size_t bytes) -> void* {
    void* p = (void*)wp;
    wp += (bytes + 255) & ~(size_t)255;
    return p;
  };
  __bf16* x_bf      = (__bf16*)arena((size_t)MLA_BS * MLA_DIM * 2);
  __bf16* wq_a_bf   = (__bf16*)arena((size_t)MLA_QLORA * MLA_DIM * 2);
  __bf16* wq_b_bf   = (__bf16*)arena((size_t)MLA_NH * MLA_QKHEAD * MLA_QLORA * 2);
  __bf16* wkv_a_bf  = (__bf16*)arena((size_t)MLA_KD * MLA_DIM * 2);
  __bf16* wkv_b_bf  = (__bf16*)arena((size_t)MLA_NH * 256 * MLA_KVLORA * 2);
  __bf16* wo_bf     = (__bf16*)arena((size_t)MLA_DIM * MLA_DIM * 2);
  float*  q_a_f     = (float*) arena((size_t)MLA_BS * MLA_QLORA * 4);
  __bf16* q_an_bf   = (__bf16*)arena((size_t)MLA_BS * MLA_QLORA * 2);
  float*  q_f       = (float*) arena((size_t)MLA_BS * MLA_NH * MLA_QKHEAD * 4);
  __bf16* q_nope_bf = (__bf16*)arena((size_t)MLA_BS * MLA_NH * MLA_NOPE * 2);
  __bf16* q_pe_bf   = (__bf16*)arena((size_t)MLA_BS * MLA_NH * MLA_ROPE * 2);
  float*  kv_f      = (float*) arena((size_t)MLA_BS * MLA_KD * 4);
  __bf16* kvc_bf    = (__bf16*)arena((size_t)MLA_BS * MLA_KVLORA * 2);
  __bf16* kpe_bf    = (__bf16*)arena((size_t)MLA_BS * MLA_ROPE * 2);
  __bf16* q_abs_bf  = (__bf16*)arena((size_t)MLA_BS * MLA_NH * MLA_KVLORA * 2);
  __bf16* o_ctx_bf  = (__bf16*)arena((size_t)MLA_BS * MLA_NH * MLA_KVLORA * 2);
  __bf16* o_v_bf    = (__bf16*)arena((size_t)MLA_BS * MLA_NH * MLA_VHEAD * 2);

  auto cvt = [&](const float* src, __bf16* dst, long n) {
    int blocks = (int)((n + 255) / 256);
    mla_cvt_bf16<<<blocks, 256, 0, stream>>>(src, dst, n);
  };
  auto gemm_k1 = [&](const __bf16* A, long lda, const __bf16* B, long ldbn,
                     float* Cf, __bf16* Cb, long ldc, int M, int N, int K) {
    long tiles = (long)(M >> 4) * (N >> 6);
    int blocks = (int)((tiles + 7) / 8);
    mla_wmma_gemm_t<true><<<blocks, 256, 0, stream>>>(A, lda, B, ldbn, 1, Cf, Cb, ldc, M, N, K);
  };
  auto gemm_ks = [&](const __bf16* A, long lda, const __bf16* B, long ldbn, long ldbk,
                     float* Cf, __bf16* Cb, long ldc, int M, int N, int K) {
    long tiles = (long)(M >> 4) * (N >> 6);
    int blocks = (int)((tiles + 7) / 8);
    mla_wmma_gemm_t<false><<<blocks, 256, 0, stream>>>(A, lda, B, ldbn, ldbk, Cf, Cb, ldc, M, N, K);
  };

  // 1) bf16 conversions (weights + x)
  cvt(x,      x_bf,     (long)MLA_BS * MLA_DIM);
  cvt(wq_a,   wq_a_bf,  (long)MLA_QLORA * MLA_DIM);
  cvt(wq_b,   wq_b_bf,  (long)MLA_NH * MLA_QKHEAD * MLA_QLORA);
  cvt(wkv_a,  wkv_a_bf, (long)MLA_KD * MLA_DIM);
  cvt(wkv_b,  wkv_b_bf, (long)MLA_NH * 256 * MLA_KVLORA);
  cvt(wo,     wo_bf,    (long)MLA_DIM * MLA_DIM);

  // 2) q_a = x @ wq_a^T ; rmsnorm
  gemm_k1(x_bf, MLA_DIM, wq_a_bf, MLA_DIM, q_a_f, nullptr, MLA_QLORA,
          MLA_BS, MLA_QLORA, MLA_DIM);
  mla_rmsnorm<<<MLA_BS, 256, 0, stream>>>(q_a_f, MLA_QLORA, q_norm_w,
                                          q_an_bf, MLA_QLORA, MLA_QLORA);

  // 3) q = q_norm @ wq_b^T ; split nope / rope(pe)
  gemm_k1(q_an_bf, MLA_QLORA, wq_b_bf, MLA_QLORA, q_f, nullptr,
          MLA_NH * MLA_QKHEAD, MLA_BS, MLA_NH * MLA_QKHEAD, MLA_QLORA);
  mla_qsplit_rope<<<MLA_BS, 256, 0, stream>>>(q_f, freqs_cos, freqs_sin,
                                              q_nope_bf, q_pe_bf);

  // 4) kv = x @ wkv_a^T ; rmsnorm(c) -> kv_cache ; rope(k_pe)
  gemm_k1(x_bf, MLA_DIM, wkv_a_bf, MLA_DIM, kv_f, nullptr, MLA_KD,
          MLA_BS, MLA_KD, MLA_DIM);
  mla_rmsnorm<<<MLA_BS, 256, 0, stream>>>(kv_f, MLA_KD, kv_norm_w,
                                          kvc_bf, MLA_KVLORA, MLA_KVLORA);
  mla_krope<<<MLA_BS, 32, 0, stream>>>(kv_f, freqs_cos, freqs_sin, kpe_bf);

  // 5) per-head absorbed Q: q_abs[:,h,:] = q_nope[:,h,:] @ wkvb[h,:128,:]
  //    (B has [K,N] memory order: ldbn=1, ldbk=512)
  for (int h = 0; h < MLA_NH; ++h) {
    gemm_ks(q_nope_bf + (long)h * MLA_NOPE, MLA_NH * MLA_NOPE,
            wkv_b_bf + (long)h * 256 * MLA_KVLORA, 1, MLA_KVLORA,
            nullptr, q_abs_bf + (long)h * MLA_KVLORA, MLA_NH * MLA_KVLORA,
            MLA_BS, MLA_KVLORA, MLA_NOPE);
  }

  // 6) flash attention over latent cache -> o_ctx
  dim3 agrid(MLA_S / 16, MLA_NH, MLA_B);
  mla_flash_attn<<<agrid, 256, 0, stream>>>(q_abs_bf, q_pe_bf, kvc_bf, kpe_bf,
                                            mask, o_ctx_bf);

  // 7) per-head V-up: o_v[:,h,:] = o_ctx[:,h,:] @ wkvb[h,128:,:]^T (row-major [N,K])
  for (int h = 0; h < MLA_NH; ++h) {
    gemm_k1(o_ctx_bf + (long)h * MLA_KVLORA, MLA_NH * MLA_KVLORA,
            wkv_b_bf + (long)h * 256 * MLA_KVLORA + (long)MLA_NOPE * MLA_KVLORA,
            MLA_KVLORA,
            nullptr, o_v_bf + (long)h * MLA_VHEAD, MLA_NH * MLA_VHEAD,
            MLA_BS, MLA_VHEAD, MLA_KVLORA);
  }

  // 8) out = o_v @ wo^T  (f32 output)
  gemm_k1(o_v_bf, MLA_DIM, wo_bf, MLA_DIM, out, nullptr, MLA_DIM,
          MLA_BS, MLA_DIM, MLA_DIM);
}
